// AllInOne_58763742544769
// MI455X (gfx1250) — matrix operand order repaired and checked by reading, not compiled, not run on hardware
//
#include <hip/hip_runtime.h>
#include <hip/hip_bf16.h>
#include <math.h>

// ---------------------------------------------------------------------------
// Types
// ---------------------------------------------------------------------------
typedef __attribute__((ext_vector_type(16))) _Float16 v16h;
typedef __attribute__((ext_vector_type(8)))  float    v8f;

union Frag  { v16h v; uint4 u[2]; _Float16 h[16]; };
union Pack8 { uint4 u; _Float16 h[8]; };

// Problem constants
#define TB   4
#define TT   1024
#define BT   (TB*TT)          // 4096
#define CEMB 512
#define C0   256
#define HEADS 8
#define DHEAD 64
#define MLPD 2048
#define NAK  7

// ---------------------------------------------------------------------------
// A-tile addressing modes (fused im2col)
//  AMODE 0: plain row-major, lda = K
//  AMODE 1: conv1 im2col: row m -> (bt, wo), data = conv0_out + (bt*26+wo)*256
//           (contiguous 3072 halfs, channel-last layout)
//  AMODE 2: conv2 im2col: row m -> (bt, fo); k -> (kt, rem) chunks of 1536,
//           source row t+kt-1 with zero pad at T edges
// ---------------------------------------------------------------------------
template<int AMODE>
__device__ __forceinline__ void load_a_group(const _Float16* __restrict__ A,
                                             int m, int koff, int K, uint4* dst) {
  if (AMODE == 0) {
    *dst = *(const uint4*)(A + (size_t)m * K + koff);
  } else if (AMODE == 1) {
    int bt = m / 15, wo = m - bt * 15;
    *dst = *(const uint4*)(A + ((size_t)bt * 26 + wo) * 256 + koff);
  } else { // AMODE == 2
    int bt = m / 3, fo = m - bt * 3;
    int kt = koff / 1536, rem = koff - kt * 1536;
    int t  = bt & (TT - 1);
    int tr = t + kt - 1;
    if (tr < 0 || tr >= TT) {
      uint4 z; z.x = z.y = z.z = z.w = 0u; *dst = z;
    } else {
      *dst = *(const uint4*)(A + ((size_t)(bt + kt - 1) * 5 + fo) * 512 + rem);
    }
  }
}

// ---------------------------------------------------------------------------
// Generic WMMA GEMM: C[M,N] = A[M,K] * B[K,N] (+ epilogue)
//  block = 256 threads (8 wave32), tile 128x128, K-step 32
//  waves arranged 4x2; each wave: 32x64 output = 2x4 v_wmma_f32_16x16x32_f16
//  A tile in LDS row-major [128][32]; B tile in LDS COLUMN-major [128 n][32 k]
//  so both A and B fragments are 2x ds_load_b128 per the ISA VGPR layouts.
//  EPI 0: Cf = acc + bias                      (f32 out)
//  EPI 1: Ch = (acc + bias) * alpha            (f16 out; q-scale)
//  EPI 2: Ch = gelu_exact(acc + bias)          (f16 out)
//  EPI 3: Cf = acc + bias + Res                (f32 out; residual)
// ---------------------------------------------------------------------------
template<int AMODE, int EPI>
__global__ __launch_bounds__(256)
void gemm_wmma(const _Float16* __restrict__ A, const _Float16* __restrict__ Bm,
               const float* __restrict__ bias, const float* __restrict__ Res,
               float* __restrict__ Cf, _Float16* __restrict__ Ch,
               int M, int N, int K, float alpha) {
  __shared__ _Float16 Asm[128 * 32];   // 8 KB, row-major [m][k]
  __shared__ _Float16 Bsm[128 * 32];   // 8 KB, col-major [n][k]

  const int tid   = threadIdx.x;
  const int lane  = tid & 31;
  const int wid   = tid >> 5;      // 0..7
  const int waveM = wid >> 1;      // 0..3  -> 32 rows each
  const int waveN = wid & 1;       // 0..1  -> 64 cols each
  const int m0    = blockIdx.y * 128;
  const int n0    = blockIdx.x * 128;
  const bool hi   = lane >= 16;
  const int  l16  = lane & 15;

  v8f acc[2][4];
  v8f vz = {};
#pragma unroll
  for (int mi = 0; mi < 2; ++mi)
#pragma unroll
    for (int ni = 0; ni < 4; ++ni) acc[mi][ni] = vz;

  for (int k0 = 0; k0 < K; k0 += 32) {
    // --- stage A tile 128x32 (512 uint4, 2/thread), row-major ---
#pragma unroll
    for (int i = 0; i < 2; ++i) {
      int u    = tid + i * 256;
      int row  = u >> 2;
      int koff = (u & 3) * 8;
      uint4 val;
      load_a_group<AMODE>(A, m0 + row, k0 + koff, K, &val);
      *(uint4*)(&Asm[row * 32 + koff]) = val;
    }
    // --- stage B tile 32x128 (512 uint4, 2/thread), transposed to [n][k] ---
#pragma unroll
    for (int i = 0; i < 2; ++i) {
      int u    = tid + i * 256;
      int row  = u >> 4;             // k within tile (16 uint4 per 128-col row)
      int coff = (u & 15) * 8;       // n within tile
      Pack8 p;
      p.u = *(const uint4*)(Bm + (size_t)(k0 + row) * N + n0 + coff);
#pragma unroll
      for (int j = 0; j < 8; ++j) Bsm[(coff + j) * 32 + row] = p.h[j];
    }
    __syncthreads();

    // --- build fragments per ISA VGPR layouts (all 2x b128 loads) ---
    Frag af[2]; Frag bf[4];
    const int i0 = hi ? 1 : 0;       // A: lanes16-31 hold K=8..15 then 24..31
#pragma unroll
    for (int mi = 0; mi < 2; ++mi) {
      int r = waveM * 32 + mi * 16 + l16;
      const uint4* rp = (const uint4*)(&Asm[r * 32]);
      af[mi].u[0] = rp[i0];
      af[mi].u[1] = rp[i0 + 2];
    }
    const int kb = hi ? 1 : 0;       // B: lanes16-31 hold K=8..15 then 24..31
#pragma unroll
    for (int ni = 0; ni < 4; ++ni) {
      int col = waveN * 64 + ni * 16 + l16;
      const uint4* cp = (const uint4*)(&Bsm[col * 32]);
      bf[ni].u[0] = cp[kb];
      bf[ni].u[1] = cp[kb + 2];
    }

#pragma unroll
    for (int mi = 0; mi < 2; ++mi)
#pragma unroll
      for (int ni = 0; ni < 4; ++ni)
        acc[mi][ni] = __builtin_amdgcn_wmma_f32_16x16x32_f16(
            false, af[mi].v, false, bf[ni].v, (short)0, acc[mi][ni],
            false, false);
    __syncthreads();
  }

  // --- epilogue ---
#pragma unroll
  for (int mi = 0; mi < 2; ++mi) {
#pragma unroll
    for (int ni = 0; ni < 4; ++ni) {
      int col = n0 + waveN * 64 + ni * 16 + l16;
      float bcol = bias[col];
#pragma unroll
      for (int r = 0; r < 8; ++r) {
        int row = m0 + waveM * 32 + mi * 16 + (hi ? 8 : 0) + r;
        float v = acc[mi][ni][r] + bcol;
        size_t idx = (size_t)row * N + col;
        if (EPI == 0) {
          Cf[idx] = v;
        } else if (EPI == 1) {
          Ch[idx] = (_Float16)(v * alpha);
        } else if (EPI == 2) {
          float g = 0.5f * v * (1.0f + erff(v * 0.70710678118654752f));
          Ch[idx] = (_Float16)g;
        } else {
          Cf[idx] = v + Res[idx];
        }
      }
    }
  }
}

// ---------------------------------------------------------------------------
// conv0 (1ch in, 3x3) + pool3 + relu, channel-last f16 out [bt][26][256]
// one block per (bt, f'), thread = out channel
// ---------------------------------------------------------------------------
__global__ __launch_bounds__(256)
void conv0_pool_relu(const float* __restrict__ x, const float* __restrict__ w,
                     const float* __restrict__ bias, _Float16* __restrict__ out) {
  __shared__ float xs[15];
  int bid = blockIdx.x;               // (b*T + t)*26 + fp
  int fp  = bid % 26;
  int bt  = bid / 26;
  int t   = bt & (TT - 1);
  int tid = threadIdx.x;
  if (tid < 15) {
    int kt = tid / 5, j = tid - kt * 5;
    int tr = t + kt - 1;
    float v = 0.0f;
    if (tr >= 0 && tr < TT) v = x[(size_t)(bt + kt - 1) * 80 + 3 * fp + j];
    xs[tid] = v;
  }
  __syncthreads();
  const float* wp = w + tid * 9;
  float w0 = wp[0], w1 = wp[1], w2 = wp[2], w3 = wp[3], w4 = wp[4],
        w5 = wp[5], w6 = wp[6], w7 = wp[7], w8 = wp[8];
  float bb = bias[tid];
  float m = -3.4e38f;
#pragma unroll
  for (int p = 0; p < 3; ++p) {
    float a = bb
      + xs[p] * w0 + xs[p + 1] * w1 + xs[p + 2] * w2
      + xs[5 + p] * w3 + xs[5 + p + 1] * w4 + xs[5 + p + 2] * w5
      + xs[10 + p] * w6 + xs[10 + p + 1] * w7 + xs[10 + p + 2] * w8;
    m = fmaxf(m, a);
  }
  out[(size_t)bid * 256 + tid] = (_Float16)fmaxf(m, 0.0f);
}

// pool3 over wo (15->5) + relu, f32 -> f16 channel-last [bt][5][512]
__global__ void pool_relu_w(const float* __restrict__ in, _Float16* __restrict__ out) {
  int id = blockIdx.x * blockDim.x + threadIdx.x;     // BT*5*512
  if (id >= BT * 5 * 512) return;
  int co = id & 511;
  int r  = id >> 9;
  int wp = r % 5;
  int bt = r / 5;
  size_t ib = ((size_t)bt * 15 + wp * 3) * 512 + co;
  float m = fmaxf(fmaxf(in[ib], in[ib + 512]), in[ib + 1024]);
  out[((size_t)bt * 5 + wp) * 512 + co] = (_Float16)fmaxf(m, 0.0f);
}

// pool3 over fo (3->1) + relu -> f32 h[bt][512]
__global__ void pool_relu_f(const float* __restrict__ in, float* __restrict__ out) {
  int id = blockIdx.x * blockDim.x + threadIdx.x;     // BT*512
  if (id >= BT * 512) return;
  int co = id & 511;
  int bt = id >> 9;
  size_t ib = (size_t)bt * 3 * 512 + co;
  float m = fmaxf(fmaxf(in[ib], in[ib + 512]), in[ib + 1024]);
  out[(size_t)bt * 512 + co] = fmaxf(m, 0.0f);
}

// ---------------------------------------------------------------------------
// LayerNorm over C=512, one block (256 threads) per row; f32 and/or f16 out
// ---------------------------------------------------------------------------
__global__ __launch_bounds__(256)
void ln_kernel(const float* __restrict__ in, const float* __restrict__ g,
               const float* __restrict__ bta, float* __restrict__ outF,
               _Float16* __restrict__ outH) {
  __shared__ float red[256];
  int row = blockIdx.x;
  int tid = threadIdx.x;
  const float* x = in + (size_t)row * CEMB;
  float a = x[tid], b = x[tid + 256];
  red[tid] = a + b;
  __syncthreads();
  for (int s = 128; s > 0; s >>= 1) {
    if (tid < s) red[tid] += red[tid + s];
    __syncthreads();
  }
  float mean = red[0] * (1.0f / CEMB);
  __syncthreads();
  float d0 = a - mean, d1 = b - mean;
  red[tid] = d0 * d0 + d1 * d1;
  __syncthreads();
  for (int s = 128; s > 0; s >>= 1) {
    if (tid < s) red[tid] += red[tid + s];
    __syncthreads();
  }
  float rstd = rsqrtf(red[0] * (1.0f / CEMB) + 1e-5f);
  float y0 = d0 * rstd * g[tid] + bta[tid];
  float y1 = d1 * rstd * g[tid + 256] + bta[tid + 256];
  size_t o = (size_t)row * CEMB;
  if (outF) { outF[o + tid] = y0; outF[o + tid + 256] = y1; }
  if (outH) { outH[o + tid] = (_Float16)y0; outH[o + tid + 256] = (_Float16)y1; }
}

// ---------------------------------------------------------------------------
// 1D neighborhood attention (K=7, D=64). One wave per (b,h,t); lane covers
// d and d+32; butterfly shfl reduce for scores; softmax + rpb in registers.
// ---------------------------------------------------------------------------
__global__ __launch_bounds__(256)
void na1d_kernel(const _Float16* __restrict__ q, const _Float16* __restrict__ kk,
                 const _Float16* __restrict__ vv, const float* __restrict__ rpb,
                 _Float16* __restrict__ ctx) {
  int gwid = (blockIdx.x * blockDim.x + threadIdx.x) >> 5;   // 0 .. B*H*T-1
  int lane = threadIdx.x & 31;
  int b   = gwid >> 13;            // / (H*T)
  int rem = gwid & 8191;
  int h   = rem >> 10;
  int t   = rem & 1023;
  int ni  = t - NAK / 2;
  if (ni < 0) ni = 0;
  if (ni > TT - NAK) ni = TT - NAK;

  size_t qoff = ((size_t)(b * TT + t)) * CEMB + h * DHEAD;
  float q0 = (float)q[qoff + lane];
  float q1 = (float)q[qoff + lane + 32];

  float sc[NAK];
#pragma unroll
  for (int j = 0; j < NAK; ++j) {
    int tr = ni + j;
    size_t ko = ((size_t)(b * TT + tr)) * CEMB + h * DHEAD;
    float s = q0 * (float)kk[ko + lane] + q1 * (float)kk[ko + lane + 32];
#pragma unroll
    for (int o = 16; o > 0; o >>= 1) s += __shfl_xor(s, o, 32);
    sc[j] = s + rpb[h * (2 * NAK - 1) + (tr - t + NAK - 1)];
  }
  float mx = sc[0];
#pragma unroll
  for (int j = 1; j < NAK; ++j) mx = fmaxf(mx, sc[j]);
  float den = 0.0f;
#pragma unroll
  for (int j = 0; j < NAK; ++j) { sc[j] = expf(sc[j] - mx); den += sc[j]; }
  float inv = 1.0f / den;
  float o0 = 0.0f, o1 = 0.0f;
#pragma unroll
  for (int j = 0; j < NAK; ++j) {
    int tr = ni + j;
    size_t vo = ((size_t)(b * TT + tr)) * CEMB + h * DHEAD;
    float p = sc[j] * inv;
    o0 += p * (float)vv[vo + lane];
    o1 += p * (float)vv[vo + lane + 32];
  }
  ctx[qoff + lane]      = (_Float16)o0;
  ctx[qoff + lane + 32] = (_Float16)o1;
}

// ---------------------------------------------------------------------------
// Weight packing / conversion kernels
// ---------------------------------------------------------------------------
// conv1_w [512][256][1][12] f32 -> B1[(kw*256+ci)][512] f16
__global__ void pack_w1(const float* __restrict__ w, _Float16* __restrict__ o) {
  int id = blockIdx.x * blockDim.x + threadIdx.x;   // 3072*512
  if (id >= 3072 * 512) return;
  int co = id & 511;
  int k  = id >> 9;
  int kw = k >> 8;
  int ci = k & 255;
  o[id] = (_Float16)w[((size_t)co * 256 + ci) * 12 + kw];
}
// conv2_w [512][512][3][3] f32 -> B2[((kt*3+kf)*512+ci)][512] f16
__global__ void pack_w2(const float* __restrict__ w, _Float16* __restrict__ o) {
  int id = blockIdx.x * blockDim.x + threadIdx.x;   // 4608*512
  if (id >= 4608 * 512) return;
  int co = id & 511;
  int k  = id >> 9;
  int kt = k / 1536;
  int r  = k - kt * 1536;
  int kf = r >> 9;
  int ci = r & 511;
  o[id] = (_Float16)w[(((size_t)co * 512 + ci) * 3 + kt) * 3 + kf];
}
// plain f32 -> f16 (row-major [K][N] kept as-is)
__global__ void cvt_f32_f16(const float* __restrict__ in, _Float16* __restrict__ o, int n) {
  int id = blockIdx.x * blockDim.x + threadIdx.x;
  if (id < n) o[id] = (_Float16)in[id];
}

// ---------------------------------------------------------------------------
// Host launch
// ---------------------------------------------------------------------------
extern "C" void kernel_launch(void* const* d_in, const int* in_sizes, int n_in,
                              void* d_out, int out_size, void* d_ws, size_t ws_size,
                              hipStream_t stream) {
  const float* x        = (const float*)d_in[0];
  const float* conv0_w  = (const float*)d_in[1];
  const float* conv0_b  = (const float*)d_in[2];
  const float* conv1_w  = (const float*)d_in[3];
  const float* conv1_b  = (const float*)d_in[4];
  const float* conv2_w  = (const float*)d_in[5];
  const float* conv2_b  = (const float*)d_in[6];
  const float* emb_ln_g = (const float*)d_in[7];
  const float* emb_ln_b = (const float*)d_in[8];
  const float* q_w      = (const float*)d_in[9];
  const float* q_b      = (const float*)d_in[10];
  const float* k_w      = (const float*)d_in[11];
  const float* k_b      = (const float*)d_in[12];
  const float* v_w      = (const float*)d_in[13];
  const float* v_b      = (const float*)d_in[14];
  const float* rpb      = (const float*)d_in[15];
  const float* ao_w     = (const float*)d_in[16];
  const float* ao_b     = (const float*)d_in[17];
  const float* ln1_g    = (const float*)d_in[18];
  const float* ln1_b    = (const float*)d_in[19];
  const float* ln2_g    = (const float*)d_in[20];
  const float* ln2_b    = (const float*)d_in[21];
  const float* m1_w     = (const float*)d_in[22];
  const float* m1_b     = (const float*)d_in[23];
  const float* m2_w     = (const float*)d_in[24];
  const float* m2_b     = (const float*)d_in[25];
  float* out = (float*)d_out;

  char* ws = (char*)d_ws;
  constexpr size_t MB = 1024ull * 1024ull;
  // Buffers (overlays are safe: strictly sequential stream order)
  _Float16* conv0_out = (_Float16*)(ws);             // 54.6 MB  [bt][26][256]
  float*    g1out     = (float*)(ws + 56 * MB);      // 125.9 MB [bt][15][512]
  _Float16* pooled1   = (_Float16*)(ws);             // overlay conv0_out (21 MB)
  float*    g2out     = (float*)(ws + 56 * MB);      // overlay g1out (25.2 MB)
  float*    hbuf      = (float*)(ws + 84 * MB);      // 8.4 MB
  float*    emb       = (float*)(ws + 93 * MB);      // 8.4 MB (shortcut)
  _Float16* ln1h      = (_Float16*)(ws + 102 * MB);
  _Float16* qb        = (_Float16*)(ws + 107 * MB);
  _Float16* kb        = (_Float16*)(ws + 112 * MB);
  _Float16* vb        = (_Float16*)(ws + 117 * MB);
  _Float16* ctx       = (_Float16*)(ws + 122 * MB);
  _Float16* ln2h      = (_Float16*)(ws + 127 * MB);
  _Float16* mid       = (_Float16*)(ws + 132 * MB);  // 16.8 MB [4096][2048]
  _Float16* w1p       = (_Float16*)(ws + 149 * MB);  // 3.15 MB
  _Float16* w2p       = (_Float16*)(ws + 153 * MB);  // 4.72 MB
  _Float16* qwh       = (_Float16*)(ws + 158 * MB);
  _Float16* kwh       = (_Float16*)(ws + 159 * MB);
  _Float16* vwh       = (_Float16*)(ws + 160 * MB);
  _Float16* owh       = (_Float16*)(ws + 161 * MB);
  _Float16* m1wh      = (_Float16*)(ws + 162 * MB);  // 2.1 MB
  _Float16* m2wh      = (_Float16*)(ws + 165 * MB);  // 2.1 MB

  // --- weight packing / conversion ---
  pack_w1<<<(3072 * 512 + 255) / 256, 256, 0, stream>>>(conv1_w, w1p);
  pack_w2<<<(4608 * 512 + 255) / 256, 256, 0, stream>>>(conv2_w, w2p);
  cvt_f32_f16<<<(512 * 512 + 255) / 256, 256, 0, stream>>>(q_w, qwh, 512 * 512);
  cvt_f32_f16<<<(512 * 512 + 255) / 256, 256, 0, stream>>>(k_w, kwh, 512 * 512);
  cvt_f32_f16<<<(512 * 512 + 255) / 256, 256, 0, stream>>>(v_w, vwh, 512 * 512);
  cvt_f32_f16<<<(512 * 512 + 255) / 256, 256, 0, stream>>>(ao_w, owh, 512 * 512);
  cvt_f32_f16<<<(512 * 2048 + 255) / 256, 256, 0, stream>>>(m1_w, m1wh, 512 * 2048);
  cvt_f32_f16<<<(2048 * 512 + 255) / 256, 256, 0, stream>>>(m2_w, m2wh, 2048 * 512);

  // --- conv stem ---
  conv0_pool_relu<<<BT * 26, 256, 0, stream>>>(x, conv0_w, conv0_b, conv0_out);

  // conv1 as fused-im2col WMMA GEMM: M=61440, K=3072, N=512
  gemm_wmma<1, 0><<<dim3(512 / 128, 61440 / 128), 256, 0, stream>>>(
      conv0_out, w1p, conv1_b, nullptr, g1out, nullptr, 61440, 512, 3072, 1.0f);
  pool_relu_w<<<(BT * 5 * 512 + 255) / 256, 256, 0, stream>>>(g1out, pooled1);

  // conv2 as fused-im2col WMMA GEMM: M=12288, K=4608, N=512
  gemm_wmma<2, 0><<<dim3(512 / 128, 12288 / 128), 256, 0, stream>>>(
      pooled1, w2p, conv2_b, nullptr, g2out, nullptr, 12288, 512, 4608, 1.0f);
  pool_relu_f<<<(BT * 512 + 255) / 256, 256, 0, stream>>>(g2out, hbuf);

  // --- embedding LN, pre-attn LN ---
  ln_kernel<<<BT, 256, 0, stream>>>(hbuf, emb_ln_g, emb_ln_b, emb, nullptr);
  ln_kernel<<<BT, 256, 0, stream>>>(emb, ln1_g, ln1_b, nullptr, ln1h);

  // --- q/k/v projections (q scaled by 1/sqrt(64)) ---
  gemm_wmma<0, 1><<<dim3(512 / 128, BT / 128), 256, 0, stream>>>(
      ln1h, qwh, q_b, nullptr, nullptr, qb, BT, 512, 512, 0.125f);
  gemm_wmma<0, 1><<<dim3(512 / 128, BT / 128), 256, 0, stream>>>(
      ln1h, kwh, k_b, nullptr, nullptr, kb, BT, 512, 512, 1.0f);
  gemm_wmma<0, 1><<<dim3(512 / 128, BT / 128), 256, 0, stream>>>(
      ln1h, vwh, v_b, nullptr, nullptr, vb, BT, 512, 512, 1.0f);

  // --- neighborhood attention ---
  na1d_kernel<<<(TB * HEADS * TT) / 8, 256, 0, stream>>>(qb, kb, vb, rpb, ctx);

  // --- out projection + residual (writes d_out = hs) ---
  gemm_wmma<0, 3><<<dim3(512 / 128, BT / 128), 256, 0, stream>>>(
      ctx, owh, ao_b, emb, out, nullptr, BT, 512, 512, 1.0f);

  // --- LN2 + MLP ---
  ln_kernel<<<BT, 256, 0, stream>>>(out, ln2_g, ln2_b, nullptr, ln2h);
  gemm_wmma<0, 2><<<dim3(2048 / 128, BT / 128), 256, 0, stream>>>(
      ln2h, m1wh, m1_b, nullptr, nullptr, mid, BT, 2048, 512, 1.0f);
  // final: d_out = hs + mlp2(mid)  (same-thread RMW of d_out as residual)
  gemm_wmma<0, 3><<<dim3(512 / 128, BT / 128), 256, 0, stream>>>(
      mid, m2wh, m2_b, out, out, nullptr, BT, 512, 2048, 1.0f);

  (void)in_sizes; (void)n_in; (void)out_size; (void)ws_size;
}
